// GraphTransformer_17892833755186
// MI455X (gfx1250) — matrix operand order repaired and checked by reading, not compiled, not run on hardware
//
#include <hip/hip_runtime.h>
#include <hip/hip_bf16.h>

typedef __attribute__((ext_vector_type(16))) __bf16 v16bf;
typedef __attribute__((ext_vector_type(8)))  float  v8f;

#define HID 128

__device__ __forceinline__ unsigned short f2bf(float f) {
    unsigned int u = __float_as_uint(f);
    unsigned int r = u + 0x7FFFu + ((u >> 16) & 1u);
    return (unsigned short)(r >> 16);
}
__device__ __forceinline__ unsigned int fmap(float f) {
    unsigned int u = __float_as_uint(f);
    return (u & 0x80000000u) ? ~u : (u | 0x80000000u);
}
__device__ __forceinline__ float funmap(unsigned int u) {
    return __uint_as_float((u & 0x80000000u) ? (u ^ 0x80000000u) : ~u);
}

__global__ void k_cvt_act(const float* __restrict__ x, unsigned short* __restrict__ x16,
                          int n4, int relu) {
    int i = blockIdx.x * blockDim.x + threadIdx.x;
    if (i >= n4) return;
    float4 v = reinterpret_cast<const float4*>(x)[i];
    if (relu) {
        v.x = fmaxf(v.x, 0.f); v.y = fmaxf(v.y, 0.f);
        v.z = fmaxf(v.z, 0.f); v.w = fmaxf(v.w, 0.f);
    }
    uint2 p;
    p.x = (unsigned)f2bf(v.x) | ((unsigned)f2bf(v.y) << 16);
    p.y = (unsigned)f2bf(v.z) | ((unsigned)f2bf(v.w) << 16);
    reinterpret_cast<uint2*>(x16)[i] = p;
}

__global__ void k_cvt_w(const float* __restrict__ Wq, const float* __restrict__ Wk,
                        const float* __restrict__ Wv, const float* __restrict__ Ws,
                        unsigned short* __restrict__ W16) {
    int i = blockIdx.x * blockDim.x + threadIdx.x;
    if (i >= 4 * HID * HID) return;
    int p = i >> 14;
    int r = i & 16383;
    int n = r >> 7;
    int k = r & 127;
    const float* W = (p == 0) ? Wq : (p == 1) ? Wk : (p == 2) ? Wv : Ws;
    W16[(size_t)p * HID * HID + (size_t)n * HID + k] = f2bf(W[(size_t)k * HID + n]);
}

__global__ void k_init(unsigned int* __restrict__ mbits, float* __restrict__ s, int n) {
    int i = blockIdx.x * blockDim.x + threadIdx.x;
    if (i >= n) return;
    mbits[i] = 0x007FFFFFu;   // fmap(-inf)
    s[i] = 0.f;
}

#define LSTR 136
__global__ void __launch_bounds__(256)
k_gemm(const unsigned short* __restrict__ X16,
       const unsigned short* __restrict__ W16,
       const float* __restrict__ bq, const float* __restrict__ bk,
       const float* __restrict__ bv, const float* __restrict__ bs,
       float* __restrict__ Q, float* __restrict__ K,
       float* __restrict__ V, float* __restrict__ OUT,
       int n) {
    __shared__ __align__(16) unsigned short lA[128 * LSTR];
    __shared__ __align__(16) unsigned short lB[128 * LSTR];

    const int t    = threadIdx.x;
    const int proj = blockIdx.y;
    const int row0 = blockIdx.x * 128;
    const int nmax = n - 1;

    // ---- async direct-to-LDS tile staging (bypasses VGPRs, tracked by ASYNCcnt) ----
    // 128x128 bf16 per tile = 2048 x 16B chunks, 8 per thread per tile.
    for (int i = 0; i < 8; ++i) {
        int c    = t + i * 256;
        int row  = c >> 4;
        int col8 = (c & 15) << 3;
        int grow = row0 + row; if (grow > nmax) grow = nmax;

        unsigned ldsA  = (unsigned)(size_t)&lA[row * LSTR + col8];
        unsigned gofsA = (unsigned)(((size_t)grow * HID + col8) * 2);   // byte offset
        asm volatile("global_load_async_to_lds_b128 %0, %1, %2"
                     :: "v"(ldsA), "v"(gofsA), "s"(X16) : "memory");

        unsigned ldsB  = (unsigned)(size_t)&lB[row * LSTR + col8];
        unsigned gofsB = (unsigned)((((size_t)proj * HID * HID) + (size_t)row * HID + col8) * 2);
        asm volatile("global_load_async_to_lds_b128 %0, %1, %2"
                     :: "v"(ldsB), "v"(gofsB), "s"(W16) : "memory");
    }
    asm volatile("s_wait_asynccnt 0" ::: "memory");
    __syncthreads();

    const int wave = t >> 5;
    const int lane = t & 31;
    const int half = lane >> 4;
    const int mr   = lane & 15;

    union AFrag { v16bf v; uint4 u[2]; };
    union BFrag { v16bf v; uint4 u[2]; };
    union CAcc  { v8f   v; float f[8]; };

    CAcc acc[8];
    #pragma unroll
    for (int nt = 0; nt < 8; ++nt)
        #pragma unroll
        for (int r = 0; r < 8; ++r) acc[nt].f[r] = 0.f;

    const int arow = wave * 16 + mr;

    #pragma unroll
    for (int kk = 0; kk < 128; kk += 32) {
        AFrag a;
        const int abase = arow * LSTR + kk + half * 8;
        a.u[0] = *reinterpret_cast<const uint4*>(&lA[abase]);
        a.u[1] = *reinterpret_cast<const uint4*>(&lA[abase + 16]);

        #pragma unroll
        for (int nt = 0; nt < 8; ++nt) {
            BFrag b;
            const int bbase = (nt * 16 + mr) * LSTR + kk + half * 8;
            b.u[0] = *reinterpret_cast<const uint4*>(&lB[bbase]);
            b.u[1] = *reinterpret_cast<const uint4*>(&lB[bbase + 16]);
            acc[nt].v = __builtin_amdgcn_wmma_f32_16x16x32_bf16(
                false, a.v, false, b.v, (short)0, acc[nt].v, false, false);
        }
    }

    const float* bias = (proj == 0) ? bq : (proj == 1) ? bk : (proj == 2) ? bv : bs;
    float* outp       = (proj == 0) ? Q  : (proj == 1) ? K  : (proj == 2) ? V  : OUT;

    #pragma unroll
    for (int nt = 0; nt < 8; ++nt) {
        const int col = nt * 16 + mr;
        const float bb = bias[col];
        #pragma unroll
        for (int r = 0; r < 8; ++r) {
            int grow = row0 + wave * 16 + half * 8 + r;
            if (grow < n) outp[(size_t)grow * HID + col] = acc[nt].f[r] + bb;
        }
    }
}

__global__ void k_alpha(const float* __restrict__ Q, const float* __restrict__ K,
                        const int* __restrict__ ei, float* __restrict__ alpha,
                        unsigned int* __restrict__ mbits, int E) {
    const int lane = threadIdx.x & 31;
    const int e = blockIdx.x * (blockDim.x >> 5) + (threadIdx.x >> 5);
    if (e >= E) return;
    const int s = ei[e];
    const int d = ei[E + e];
    const float4 qv = *reinterpret_cast<const float4*>(Q + (size_t)d * HID + lane * 4);
    const float4 kv = *reinterpret_cast<const float4*>(K + (size_t)s * HID + lane * 4);
    float p = qv.x * kv.x + qv.y * kv.y + qv.z * kv.z + qv.w * kv.w;
    #pragma unroll
    for (int off = 16; off > 0; off >>= 1) p += __shfl_xor(p, off, 32);
    if (lane == 0) {
        float a = p * 0.08838834764831845f;
        alpha[e] = a;
        atomicMax(mbits + d, fmap(a));
    }
}

__global__ void k_expsum(float* __restrict__ alpha, const int* __restrict__ ei,
                         const unsigned int* __restrict__ mbits,
                         float* __restrict__ S, int E) {
    int e = blockIdx.x * blockDim.x + threadIdx.x;
    if (e >= E) return;
    int d = ei[E + e];
    float ev = __expf(alpha[e] - funmap(mbits[d]));
    alpha[e] = ev;
    atomicAdd(S + d, ev);
}

__global__ void k_scatter(const float* __restrict__ alpha, const float* __restrict__ S,
                          const float* __restrict__ V, const int* __restrict__ ei,
                          float* __restrict__ OUT, int E) {
    const int lane = threadIdx.x & 31;
    const int e = blockIdx.x * (blockDim.x >> 5) + (threadIdx.x >> 5);
    if (e >= E) return;
    const int s = ei[e];
    const int d = ei[E + e];
    const float a = alpha[e] / (S[d] + 1e-16f);
    const float4 vv = *reinterpret_cast<const float4*>(V + (size_t)s * HID + lane * 4);
    float* o = OUT + (size_t)d * HID + lane * 4;
    atomicAdd(o + 0, a * vv.x);
    atomicAdd(o + 1, a * vv.y);
    atomicAdd(o + 2, a * vv.z);
    atomicAdd(o + 3, a * vv.w);
}

static void run_layer(const float* Xin, int relu_in, const int* ei,
                      const float* Wq, const float* bq, const float* Wk, const float* bk,
                      const float* Wv, const float* bv, const float* Ws, const float* bs,
                      float* Yout,
                      unsigned short* X16, unsigned short* W16,
                      float* Q, float* K, float* V,
                      float* ALPHA, unsigned int* MB, float* S,
                      int n, int E, hipStream_t stream) {
    int n4 = n * HID / 4;
    k_cvt_act<<<(n4 + 255) / 256, 256, 0, stream>>>(Xin, X16, n4, relu_in);
    k_cvt_w<<<(4 * HID * HID + 255) / 256, 256, 0, stream>>>(Wq, Wk, Wv, Ws, W16);
    k_init<<<(n + 255) / 256, 256, 0, stream>>>(MB, S, n);

    dim3 ggrid((n + 127) / 128, 4);
    k_gemm<<<ggrid, 256, 0, stream>>>(X16, W16, bq, bk, bv, bs, Q, K, V, Yout, n);

    int eblk = (E + 7) / 8;
    k_alpha<<<eblk, 256, 0, stream>>>(Q, K, ei, ALPHA, MB, E);
    k_expsum<<<(E + 255) / 256, 256, 0, stream>>>(ALPHA, ei, MB, S, E);
    k_scatter<<<eblk, 256, 0, stream>>>(ALPHA, S, V, ei, Yout, E);
}

extern "C" void kernel_launch(void* const* d_in, const int* in_sizes, int n_in,
                              void* d_out, int out_size, void* d_ws, size_t ws_size,
                              hipStream_t stream) {
    const float* x  = (const float*)d_in[0];
    const int*   ei = (const int*)d_in[1];
    const int n = in_sizes[0] / HID;
    const int E = in_sizes[1] / 2;

    // JAX tree-flatten (sorted keys): conv1{k,q,skip,v}{W,b}, conv2 same
    const float* c1kW = (const float*)d_in[2];  const float* c1kb = (const float*)d_in[3];
    const float* c1qW = (const float*)d_in[4];  const float* c1qb = (const float*)d_in[5];
    const float* c1sW = (const float*)d_in[6];  const float* c1sb = (const float*)d_in[7];
    const float* c1vW = (const float*)d_in[8];  const float* c1vb = (const float*)d_in[9];
    const float* c2kW = (const float*)d_in[10]; const float* c2kb = (const float*)d_in[11];
    const float* c2qW = (const float*)d_in[12]; const float* c2qb = (const float*)d_in[13];
    const float* c2sW = (const float*)d_in[14]; const float* c2sb = (const float*)d_in[15];
    const float* c2vW = (const float*)d_in[16]; const float* c2vb = (const float*)d_in[17];

    char* ws = (char*)d_ws;
    size_t off = 0;
    unsigned short* X16 = (unsigned short*)(ws + off); off += (size_t)n * HID * 2;
    off = (off + 255) & ~(size_t)255;
    unsigned short* W16 = (unsigned short*)(ws + off); off += (size_t)4 * HID * HID * 2;
    off = (off + 255) & ~(size_t)255;
    float* Q  = (float*)(ws + off); off += (size_t)n * HID * 4;
    float* K  = (float*)(ws + off); off += (size_t)n * HID * 4;
    float* V  = (float*)(ws + off); off += (size_t)n * HID * 4;
    float* H  = (float*)(ws + off); off += (size_t)n * HID * 4;
    float* ALPHA = (float*)(ws + off); off += (size_t)E * 4;
    unsigned int* MB = (unsigned int*)(ws + off); off += (size_t)n * 4;
    float* S  = (float*)(ws + off); off += (size_t)n * 4;

    float* out = (float*)d_out;

    run_layer(x, 0, ei, c1qW, c1qb, c1kW, c1kb, c1vW, c1vb, c1sW, c1sb,
              H, X16, W16, Q, K, V, ALPHA, MB, S, n, E, stream);
    run_layer(H, 1, ei, c2qW, c2qb, c2kW, c2kb, c2vW, c2vb, c2sW, c2sb,
              out, X16, W16, Q, K, V, ALPHA, MB, S, n, E, stream);
}
